// Self_Attn_11811160064198
// MI455X (gfx1250) — matrix operand order repaired and checked by reading, not compiled, not run on hardware
//
#include <hip/hip_runtime.h>

typedef __attribute__((ext_vector_type(16))) _Float16 v16h;
typedef __attribute__((ext_vector_type(8)))  float    v8f;

#define B_  4
#define C_  128
#define C8_ 16
#define N_  4096

// ---------------------------------------------------------------------------
// Kernel 1: 1x1x1 conv == per-voxel linear. Produce qT/kT [B,N,16] f16 and
// v [B,C,N] f16 (bias included). Block: (b, 32 voxels), x tile in LDS.
// Also writes a 16-half zero pad used by energy_kernel's B-operand hi lanes.
// ---------------------------------------------------------------------------
__global__ void qkv_kernel(const float* __restrict__ x,
                           const float* __restrict__ Wq, const float* __restrict__ bq,
                           const float* __restrict__ Wk, const float* __restrict__ bk,
                           const float* __restrict__ Wv, const float* __restrict__ bv,
                           _Float16* __restrict__ qT, _Float16* __restrict__ kT,
                           _Float16* __restrict__ vv, _Float16* __restrict__ zpad) {
  __shared__ float xs[C_ * 32];
  const int b   = blockIdx.x;
  const int n0  = blockIdx.y * 32;
  const int tid = threadIdx.x;
  if (b == 0 && blockIdx.y == 0 && tid < 16) zpad[tid] = (_Float16)0.0f;
  for (int idx = tid; idx < C_ * 32; idx += 256) {
    const int c = idx >> 5, n = idx & 31;
    xs[idx] = x[((size_t)(b * C_ + c)) * N_ + n0 + n];
  }
  __syncthreads();
  // 160 output channels (16 q + 16 k + 128 v) x 32 voxels = 5120 dots / block
  for (int idx = tid; idx < 160 * 32; idx += 256) {
    const int n = idx & 31, oo = idx >> 5;
    const float* wrow; float acc;
    if (oo < 16)      { wrow = Wq + oo * C_;        acc = bq[oo];      }
    else if (oo < 32) { wrow = Wk + (oo - 16) * C_; acc = bk[oo - 16]; }
    else              { wrow = Wv + (oo - 32) * C_; acc = bv[oo - 32]; }
    #pragma unroll 8
    for (int c = 0; c < C_; ++c) acc += wrow[c] * xs[c * 32 + n];
    const int gn = n0 + n;
    if (oo < 16)      qT[((size_t)(b * N_ + gn)) * C8_ + oo]          = (_Float16)acc;
    else if (oo < 32) kT[((size_t)(b * N_ + gn)) * C8_ + (oo - 16)]   = (_Float16)acc;
    else              vv[((size_t)(b * C_ + (oo - 32))) * N_ + gn]    = (_Float16)acc;
  }
}

// ---------------------------------------------------------------------------
// Kernel 2: energy[b,i,j] = sum_o q[b,o,i]*k[b,o,j].  WMMA f32_16x16x32_f16,
// K=16 padded to 32 with zeros. Block = (b, i-tile), 8 waves sweep j-tiles.
// A lanes 0-15 hold K 0-15 (qT row), lanes 16-31 hold K 16-31 = 0 (zero pad).
// B lanes 0-15 hold K 0-15 (kT row, one 32B v16h load); lanes 16-31 hold
// K 16-31 -> read from a 32B zero pad (uniform load, no EXEC divergence).
// ---------------------------------------------------------------------------
__global__ void energy_kernel(const _Float16* __restrict__ qT,
                              const _Float16* __restrict__ kT,
                              const _Float16* __restrict__ zpad,
                              float* __restrict__ att) {
  const int b    = blockIdx.x;
  const int i0   = blockIdx.y * 16;
  const int tid  = threadIdx.x;
  const int wave = tid >> 5, lane = tid & 31;
  const int nl   = lane & 15, hi = lane >> 4;

  v16h a = {};
  const _Float16* qrow = qT + ((size_t)(b * N_ + i0 + nl)) * C8_;
  #pragma unroll
  for (int t = 0; t < 16; ++t) a[t] = hi ? (_Float16)0.0f : qrow[t];   // hi lanes stay 0 (K pad)

  float* eb = att + (size_t)b * N_ * N_;
  const _Float16* kbase = kT + (size_t)b * N_ * C8_;
  for (int jt = wave; jt < N_ / 16; jt += 8) {
    const int j0 = jt * 16;
    const _Float16* ksrc = hi ? zpad : (kbase + (size_t)(j0 + nl) * C8_);
    v16h bm = *(const v16h*)ksrc;
    v8f acc = {};
    acc = __builtin_amdgcn_wmma_f32_16x16x32_f16(false, a, false, bm,
                                                 (short)0, acc, false, false);
    #pragma unroll
    for (int r = 0; r < 8; ++r)   // D: VGPR r -> row r + 8*hi, col = lane&15
      eb[(size_t)(i0 + r + 8 * hi) * N_ + j0 + nl] = acc[r];
  }
}

// ---------------------------------------------------------------------------
// Kernel 3: in-place row softmax over 4096 keys. One block per (b,i) row.
// ---------------------------------------------------------------------------
__global__ void softmax_kernel(float* __restrict__ att) {
  __shared__ float red[256];
  const int tid = threadIdx.x;
  float* row = att + (size_t)blockIdx.x * N_;
  float vals[16];
  float mx = -3.4e38f;
  #pragma unroll
  for (int t = 0; t < 16; ++t) { vals[t] = row[t * 256 + tid]; mx = fmaxf(mx, vals[t]); }
  red[tid] = mx; __syncthreads();
  for (int s = 128; s > 0; s >>= 1) {
    if (tid < s) red[tid] = fmaxf(red[tid], red[tid + s]);
    __syncthreads();
  }
  mx = red[0]; __syncthreads();
  float sum = 0.f;
  #pragma unroll
  for (int t = 0; t < 16; ++t) { vals[t] = __expf(vals[t] - mx); sum += vals[t]; }
  red[tid] = sum; __syncthreads();
  for (int s = 128; s > 0; s >>= 1) {
    if (tid < s) red[tid] += red[tid + s];
    __syncthreads();
  }
  const float inv = 1.0f / red[0];
  #pragma unroll
  for (int t = 0; t < 16; ++t) row[t * 256 + tid] = vals[t] * inv;
}

// ---------------------------------------------------------------------------
// Kernel 4: out[b,c,i] = gamma * sum_j v[b,c,j]*att[b,i,j] + x[b,c,i].
// Block = (b, i-tile); 8 waves = 8 c-tiles (full C=128).
// The 16x32 fp32 attention chunk is DMA'd into LDS with
// GLOBAL_LOAD_ASYNC_TO_LDS_B64 (ASYNCcnt), double-buffered so the copy of
// chunk k+1 overlaps the WMMA on chunk k. Async loads complete in order, so
// s_wait_asynccnt 1 after issuing the next chunk guarantees the current one
// has landed. fp32->f16 conversion happens at A-operand build from LDS.
// ---------------------------------------------------------------------------
__global__ void av_kernel(const float* __restrict__ att,
                          const _Float16* __restrict__ vv,
                          const float* __restrict__ x,
                          const float* __restrict__ gamma,
                          float* __restrict__ out) {
  __shared__ float attS[2][16 * 32];   // double-buffered chunks, 2 x 2 KB
  const int b    = blockIdx.x;
  const int i0   = blockIdx.y * 16;
  const int tid  = threadIdx.x;
  const int wave = tid >> 5, lane = tid & 31;
  const int nl   = lane & 15, hi = lane >> 4;
  const int c0   = wave * 16;

  const float*    ap = att + ((size_t)(b * N_ + i0)) * N_;
  const _Float16* vb = vv  + (size_t)b * C_ * N_;

  // per-thread async slot: 2 consecutive floats of the 16x32 chunk
  const int r2 = tid >> 4;            // row 0..15
  const int c2 = (tid & 15) * 2;      // col 0,2,...,30
  const float*   gsrc  = ap + (size_t)r2 * N_ + c2;
  const unsigned ldsEl = (unsigned)(r2 * 32 + c2) * 4u;  // byte offset, attS @ LDS 0

  // stage chunk 0 into buffer 0
  asm volatile("global_load_async_to_lds_b64 %0, %1, off"
               :: "v"(ldsEl), "v"(gsrc) : "memory");

  v8f acc = {};
  for (int kt = 0; kt < N_ / 32; ++kt) {
    const int cur = kt & 1;
    if (kt + 1 < N_ / 32) {          // stage next chunk into other buffer
      const unsigned ldsn = (unsigned)((cur ^ 1) * 2048) + ldsEl;
      const float*   gn   = gsrc + (kt + 1) * 32;
      asm volatile("global_load_async_to_lds_b64 %0, %1, off"
                   :: "v"(ldsn), "v"(gn) : "memory");
      asm volatile("s_wait_asynccnt 0x1" ::: "memory");  // current chunk done
    } else {
      asm volatile("s_wait_asynccnt 0x0" ::: "memory");
    }
    __syncthreads();                 // chunk visible to all 8 waves

    // A operand (contiguous k, same order as B): row nl, K kb..kb+15
    const float* as = &attS[cur][0];
    const int kb = hi ? 16 : 0;
    const float4 f0 = *(const float4*)(as + nl * 32 + kb);
    const float4 f1 = *(const float4*)(as + nl * 32 + kb + 4);
    const float4 f2 = *(const float4*)(as + nl * 32 + kb + 8);
    const float4 f3 = *(const float4*)(as + nl * 32 + kb + 12);
    v16h a = {(_Float16)f0.x, (_Float16)f0.y, (_Float16)f0.z, (_Float16)f0.w,
              (_Float16)f1.x, (_Float16)f1.y, (_Float16)f1.z, (_Float16)f1.w,
              (_Float16)f2.x, (_Float16)f2.y, (_Float16)f2.z, (_Float16)f2.w,
              (_Float16)f3.x, (_Float16)f3.y, (_Float16)f3.z, (_Float16)f3.w};

    // B operand: lane = column c0+nl, 16 consecutive j -> one 32B v16h load
    const _Float16* vrow = vb + (size_t)(c0 + nl) * N_ + kt * 32 + (hi ? 16 : 0);
    v16h bm = *(const v16h*)vrow;
    if (kt + 1 < N_ / 32)
      __builtin_prefetch(vrow + 32, 0, 0);   // global_prefetch_b8, next chunk

    acc = __builtin_amdgcn_wmma_f32_16x16x32_f16(false, a, false, bm,
                                                 (short)0, acc, false, false);
    __syncthreads();                 // all waves done reading this buffer
  }

  const float g = gamma[0];
  #pragma unroll
  for (int r = 0; r < 8; ++r) {
    const int i = i0 + r + 8 * hi;
    const int c = c0 + nl;
    const size_t idx = ((size_t)(b * C_ + c)) * N_ + i;
    out[idx] = g * acc[r] + x[idx];
  }
}

// ---------------------------------------------------------------------------
extern "C" void kernel_launch(void* const* d_in, const int* in_sizes, int n_in,
                              void* d_out, int out_size, void* d_ws, size_t ws_size,
                              hipStream_t stream) {
  const float* x     = (const float*)d_in[0];
  const float* Wq    = (const float*)d_in[1];
  const float* bq    = (const float*)d_in[2];
  const float* Wk    = (const float*)d_in[3];
  const float* bk    = (const float*)d_in[4];
  const float* Wv    = (const float*)d_in[5];
  const float* bv    = (const float*)d_in[6];
  const float* gamma = (const float*)d_in[7];

  float* out = (float*)d_out;                       // [B,C,D,H,W] = B*C*N floats
  float* att = out + (size_t)B_ * C_ * N_;          // [B,N,N] floats

  _Float16* qT   = (_Float16*)d_ws;                 // [B,N,16]   0.5 MB
  _Float16* kT   = qT + (size_t)B_ * N_ * C8_;      // [B,N,16]   0.5 MB
  _Float16* vv   = kT + (size_t)B_ * N_ * C8_;      // [B,C,N]    4 MB
  _Float16* zpad = vv + (size_t)B_ * C_ * N_;       // 16 halves, 32B-aligned

  hipLaunchKernelGGL(qkv_kernel,    dim3(B_, N_ / 32), dim3(256), 0, stream,
                     x, Wq, bq, Wk, bk, Wv, bv, qT, kT, vv, zpad);
  hipLaunchKernelGGL(energy_kernel, dim3(B_, N_ / 16), dim3(256), 0, stream,
                     qT, kT, zpad, att);
  hipLaunchKernelGGL(softmax_kernel, dim3(B_ * N_),    dim3(256), 0, stream,
                     att);
  hipLaunchKernelGGL(av_kernel,     dim3(B_, N_ / 16), dim3(256), 0, stream,
                     att, vv, x, gamma, out);
}
